// DeepSeekMLA_63788854280367
// MI455X (gfx1250) — matrix operand order repaired and checked by reading, not compiled, not run on hardware
//
#include <hip/hip_runtime.h>

// ---------------------------------------------------------------------------
// Problem constants (DeepSeek-style MLA forward)
// ---------------------------------------------------------------------------
#define B_DIM   2
#define T_DIM   2048
#define C_DIM   2048
#define H_NUM   16
#define HEADD   128            // NOPE(64) + ROPE(64)
#define VDIM    128
#define QRANK   1536
#define KVRANK  512
#define MTOT    (B_DIM * T_DIM)   // 4096 token rows

typedef __attribute__((ext_vector_type(16))) __bf16 v16bf;
typedef __attribute__((ext_vector_type(8)))  float  v8f;

__device__ __forceinline__ unsigned short f32_to_bf16(float f) {
  unsigned int u = __float_as_uint(f);
  unsigned int r = 0x7FFFu + ((u >> 16) & 1u);   // round-to-nearest-even
  return (unsigned short)((u + r) >> 16);
}
__device__ __forceinline__ float bf16_to_f32(unsigned short h) {
  return __uint_as_float(((unsigned int)h) << 16);
}
__device__ __forceinline__ v16bf frag_from(uint4 lo, uint4 hi) {
  union { uint4 q[2]; v16bf v; } u;
  u.q[0] = lo; u.q[1] = hi;
  return u.v;
}
// CDNA5 async copy: 16 bytes/lane global -> LDS, tracked by ASYNCcnt.
__device__ __forceinline__ void async_copy16(unsigned lds_addr, const void* gptr) {
  asm volatile("global_load_async_to_lds_b128 %0, %1, off"
               :: "v"(lds_addr), "v"((unsigned long long)(size_t)gptr)
               : "memory");
}
__device__ __forceinline__ void wait_async0() {
  asm volatile("s_wait_asynccnt 0" ::: "memory");
}

// ---------------------------------------------------------------------------
// f32 -> bf16 conversion
// ---------------------------------------------------------------------------
__global__ __launch_bounds__(256) void cvt_f32_bf16(const float* __restrict__ in,
                                                    unsigned short* __restrict__ out,
                                                    int n) {
  int i = blockIdx.x * 256 + threadIdx.x;
  if (i < n) out[i] = f32_to_bf16(in[i]);
}

// ---------------------------------------------------------------------------
// Tiled WMMA GEMM: C[M,N] = A[M,K] * W[N,K]^T   (bf16 in, f32 accumulate)
// Block = 256 threads (8 waves), block tile 128x64, BLK_K = 32.
// Wave w: rows (w&3)*32, cols (w>>2)*32 -> 2x2 grid of 16x16 accumulators.
// K-tiles staged in LDS with GLOBAL_LOAD_ASYNC_TO_LDS_B128, double-buffered.
// ---------------------------------------------------------------------------
__global__ __launch_bounds__(256) void gemm_bf16_wmma(
    const unsigned short* __restrict__ A,
    const unsigned short* __restrict__ W,
    float* __restrict__ Cf,              // f32 output (or null)
    unsigned short* __restrict__ Cb,     // bf16 output (or null)
    int M, int N, int K)
{
  __shared__ __align__(16) unsigned short As[2][128][32];   // 16 KB
  __shared__ __align__(16) unsigned short Ws[2][64][32];    //  8 KB

  const int tid  = threadIdx.x;
  const int lane = tid & 31;
  const int wave = tid >> 5;
  const int m0 = blockIdx.y * 128;
  const int n0 = blockIdx.x * 64;
  const int wm = (wave & 3) * 32;
  const int wn = (wave >> 2) * 32;
  const int r    = tid >> 2;          // 0..63 : loader row
  const int lc   = (tid & 3) * 8;     // 8 bf16 (16B) per thread per row
  const int idx  = lane & 15;
  const int hi   = lane >> 4;
  const int koff = hi * 8;            // A-fragment K base
  const int hoff = hi * 16;           // unused; B fragments also use koff

  v8f acc00 = {}, acc01 = {}, acc10 = {}, acc11 = {};

  const unsigned short* aPtr0 = A + (size_t)(m0 + r) * K + lc;
  const unsigned short* aPtr1 = A + (size_t)(m0 + 64 + r) * K + lc;
  const unsigned short* wPtr  = W + (size_t)(n0 + r) * K + lc;
  const unsigned asA0 = (unsigned)(size_t)&As[0][r][lc];       // flat->LDS: low 32b
  const unsigned asA1 = (unsigned)(size_t)&As[0][64 + r][lc];
  const unsigned asW  = (unsigned)(size_t)&Ws[0][r][lc];
  const unsigned strA = 128 * 32 * 2;   // per-buffer stride (bytes)
  const unsigned strW = 64 * 32 * 2;

  // prologue: stage tile 0 into buffer 0
  async_copy16(asA0, aPtr0);
  async_copy16(asA1, aPtr1);
  async_copy16(asW,  wPtr);
  wait_async0();
  __syncthreads();

  const int nk = K >> 5;
  int buf = 0;
  for (int kt = 0; kt < nk; ++kt) {
    if (kt + 1 < nk) {                      // prefetch next K-tile (async DMA)
      const size_t ko = (size_t)(kt + 1) * 32;
      const unsigned bo = (unsigned)(buf ^ 1);
      async_copy16(asA0 + bo * strA, aPtr0 + ko);
      async_copy16(asA1 + bo * strA, aPtr1 + ko);
      async_copy16(asW  + bo * strW, wPtr  + ko);
    }
    const unsigned short (*Ab)[32] = As[buf];
    const unsigned short (*Wb)[32] = Ws[buf];
    v16bf af0 = frag_from(*(const uint4*)&Ab[wm + idx][koff],
                          *(const uint4*)&Ab[wm + idx][koff + 16]);
    v16bf af1 = frag_from(*(const uint4*)&Ab[wm + 16 + idx][koff],
                          *(const uint4*)&Ab[wm + 16 + idx][koff + 16]);
    v16bf b0  = frag_from(*(const uint4*)&Wb[wn + idx][koff],
                          *(const uint4*)&Wb[wn + idx][koff + 16]);
    v16bf b1  = frag_from(*(const uint4*)&Wb[wn + 16 + idx][koff],
                          *(const uint4*)&Wb[wn + 16 + idx][koff + 16]);
    acc00 = __builtin_amdgcn_wmma_f32_16x16x32_bf16(false, af0, false, b0,
                                                    (short)0, acc00, false, false);
    acc01 = __builtin_amdgcn_wmma_f32_16x16x32_bf16(false, af0, false, b1,
                                                    (short)0, acc01, false, false);
    acc10 = __builtin_amdgcn_wmma_f32_16x16x32_bf16(false, af1, false, b0,
                                                    (short)0, acc10, false, false);
    acc11 = __builtin_amdgcn_wmma_f32_16x16x32_bf16(false, af1, false, b1,
                                                    (short)0, acc11, false, false);
    wait_async0();       // own prefetch landed in LDS
    __syncthreads();     // everyone done reading buf / writing buf^1
    buf ^= 1;
  }

  const int r0 = m0 + wm + hi * 8;
  const int c0 = n0 + wn + idx;
  if (Cf) {
    for (int j = 0; j < 8; ++j) {
      Cf[(size_t)(r0 + j) * N + c0]           = acc00[j];
      Cf[(size_t)(r0 + j) * N + c0 + 16]      = acc01[j];
      Cf[(size_t)(r0 + 16 + j) * N + c0]      = acc10[j];
      Cf[(size_t)(r0 + 16 + j) * N + c0 + 16] = acc11[j];
    }
  } else {
    for (int j = 0; j < 8; ++j) {
      Cb[(size_t)(r0 + j) * N + c0]           = f32_to_bf16(acc00[j]);
      Cb[(size_t)(r0 + j) * N + c0 + 16]      = f32_to_bf16(acc01[j]);
      Cb[(size_t)(r0 + 16 + j) * N + c0]      = f32_to_bf16(acc10[j]);
      Cb[(size_t)(r0 + 16 + j) * N + c0 + 16] = f32_to_bf16(acc11[j]);
    }
  }
}

// ---------------------------------------------------------------------------
// RoPE + per-head RMS-norm; splits kv into K (row-major) and V (transposed).
// One wave per (b, t, h); lane owns dims {lane, lane+32, lane+64, lane+96}.
// ---------------------------------------------------------------------------
__global__ __launch_bounds__(256) void rope_rms_split(
    const unsigned short* __restrict__ Qraw,   // [MTOT][H*128] bf16
    const unsigned short* __restrict__ KVraw,  // [MTOT][H*256] bf16
    const float* __restrict__ Cosv,            // [T][32]
    const float* __restrict__ Sinv,            // [T][32]
    unsigned short* __restrict__ Qa,           // [B*H][T][128]
    unsigned short* __restrict__ Ka,           // [B*H][T][128]
    unsigned short* __restrict__ Vt)           // [B*H][128][T]
{
  const int lane = threadIdx.x & 31;
  const int wave = threadIdx.x >> 5;
  const int g  = blockIdx.x * 8 + wave;   // ((b*T)+t)*16 + h
  const int h  = g & 15;
  const int bt = g >> 4;
  const int b  = bt / T_DIM;
  const int t  = bt - b * T_DIM;
  const float c = Cosv[t * 32 + lane];
  const float s = Sinv[t * 32 + lane];
  const float eps = 1.1920929e-7f;

  { // ---- Q head ----
    const unsigned short* p = Qraw + (size_t)bt * (H_NUM * HEADD) + h * HEADD;
    float f0 = bf16_to_f32(p[lane]);
    float f1 = bf16_to_f32(p[lane + 32]);
    float x1 = bf16_to_f32(p[lane + 64]);
    float x2 = bf16_to_f32(p[lane + 96]);
    float f2 =  x1 * c + x2 * s;
    float f3 = -x1 * s + x2 * c;
    float ss = f0 * f0 + f1 * f1 + f2 * f2 + f3 * f3;
    for (int o = 16; o; o >>= 1) ss += __shfl_xor(ss, o, 32);
    float rn = rsqrtf(ss * (1.0f / 128.0f) + eps);
    unsigned short* q = Qa + ((size_t)(b * H_NUM + h) * T_DIM + t) * HEADD;
    q[lane]      = f32_to_bf16(f0 * rn);
    q[lane + 32] = f32_to_bf16(f1 * rn);
    q[lane + 64] = f32_to_bf16(f2 * rn);
    q[lane + 96] = f32_to_bf16(f3 * rn);
  }
  { // ---- K head + V split ----
    const unsigned short* p = KVraw + (size_t)bt * (H_NUM * 256) + h * 256;
    float f0 = bf16_to_f32(p[lane]);
    float f1 = bf16_to_f32(p[lane + 32]);
    float x1 = bf16_to_f32(p[lane + 64]);
    float x2 = bf16_to_f32(p[lane + 96]);
    float f2 =  x1 * c + x2 * s;
    float f3 = -x1 * s + x2 * c;
    float ss = f0 * f0 + f1 * f1 + f2 * f2 + f3 * f3;
    for (int o = 16; o; o >>= 1) ss += __shfl_xor(ss, o, 32);
    float rn = rsqrtf(ss * (1.0f / 128.0f) + eps);
    unsigned short* k = Ka + ((size_t)(b * H_NUM + h) * T_DIM + t) * HEADD;
    k[lane]      = f32_to_bf16(f0 * rn);
    k[lane + 32] = f32_to_bf16(f1 * rn);
    k[lane + 64] = f32_to_bf16(f2 * rn);
    k[lane + 96] = f32_to_bf16(f3 * rn);
    unsigned short* v = Vt + (size_t)(b * H_NUM + h) * 128 * T_DIM;
    for (int i = 0; i < 4; ++i) {
      int d = lane + 32 * i;
      v[(size_t)d * T_DIM + t] = p[128 + d];   // already bf16
    }
  }
}

// ---------------------------------------------------------------------------
// Flash-style causal attention. 128 threads = 4 waves; each wave owns one
// 16-row Q tile, streams keys 32 at a time: QK^T (8 WMMA) -> online softmax
// (shfl row reductions) -> P transposed via per-wave LDS -> P*V (8 WMMA).
// ---------------------------------------------------------------------------
__global__ __launch_bounds__(128) void mla_flash_attn(
    const unsigned short* __restrict__ Qa,   // [B*H][T][128]
    const unsigned short* __restrict__ Ka,   // [B*H][T][128]
    const unsigned short* __restrict__ Vt,   // [B*H][128][T]
    unsigned short* __restrict__ Y)          // [B*T][H*128]
{
  __shared__ __align__(16) unsigned short Ps[4][16][32];

  const int lane = threadIdx.x & 31;
  const int wave = threadIdx.x >> 5;
  const int bh = blockIdx.y;
  const int b  = bh >> 4;
  const int h  = bh & 15;
  const int q0 = (blockIdx.x * 4 + wave) * 16;
  const int idx  = lane & 15;
  const int hi   = lane >> 4;
  const int koff = hi * 8;
  const int hoff = hi * 16;

  const unsigned short* qp = Qa + ((size_t)bh * T_DIM + q0) * HEADD;
  const unsigned short* kp = Ka + (size_t)bh * T_DIM * HEADD;
  const unsigned short* vp = Vt + (size_t)bh * 128 * T_DIM;

  v16bf qa[4];
  for (int cc = 0; cc < 4; ++cc) {
    const unsigned short* p = qp + (size_t)idx * HEADD + cc * 32;
    qa[cc] = frag_from(*(const uint4*)(p + koff), *(const uint4*)(p + koff + 16));
  }

  v8f zero = {};
  v8f acc[8];
  float mrow[8], lrow[8];
  for (int vt = 0; vt < 8; ++vt) acc[vt] = zero;
  for (int j = 0; j < 8; ++j) { mrow[j] = -3.0e38f; lrow[j] = 0.0f; }

  const float scale = 0.088388347648318447f;   // 1/sqrt(128)
  const int kend = q0 + 16;
  for (int kc = 0; kc < kend; kc += 32) {
    v8f s0 = zero, s1 = zero;
    for (int cc = 0; cc < 4; ++cc) {
      const unsigned short* k0 = kp + (size_t)(kc + idx) * HEADD + cc * 32 + koff;
      v16bf kb0 = frag_from(*(const uint4*)k0, *(const uint4*)(k0 + 16));
      s0 = __builtin_amdgcn_wmma_f32_16x16x32_bf16(false, qa[cc], false, kb0,
                                                   (short)0, s0, false, false);
      const unsigned short* k1 = kp + (size_t)(kc + 16 + idx) * HEADD + cc * 32 + koff;
      v16bf kb1 = frag_from(*(const uint4*)k1, *(const uint4*)(k1 + 16));
      s1 = __builtin_amdgcn_wmma_f32_16x16x32_bf16(false, qa[cc], false, kb1,
                                                   (short)0, s1, false, false);
    }
    // online softmax, one row per accumulator element
    for (int j = 0; j < 8; ++j) {
      const int trow = q0 + hi * 8 + j;
      float v0 = (kc + idx)      <= trow ? s0[j] * scale : -3.0e38f;
      float v1 = (kc + 16 + idx) <= trow ? s1[j] * scale : -3.0e38f;
      float mx = fmaxf(v0, v1);
      for (int o = 8; o; o >>= 1) mx = fmaxf(mx, __shfl_xor(mx, o, 16));
      float mnew  = fmaxf(mrow[j], mx);
      float alpha = __expf(mrow[j] - mnew);
      float e0 = __expf(v0 - mnew);
      float e1 = __expf(v1 - mnew);
      float sum = e0 + e1;
      for (int o = 8; o; o >>= 1) sum += __shfl_xor(sum, o, 16);
      lrow[j] = lrow[j] * alpha + sum;
      mrow[j] = mnew;
      for (int vt = 0; vt < 8; ++vt) acc[vt][j] *= alpha;
      Ps[wave][hi * 8 + j][idx]      = f32_to_bf16(e0);
      Ps[wave][hi * 8 + j][idx + 16] = f32_to_bf16(e1);
    }
    asm volatile("s_wait_dscnt 0" ::: "memory");   // LDS transpose visibility
    const unsigned short* pp = &Ps[wave][idx][0];
    v16bf pa = frag_from(*(const uint4*)(pp + koff), *(const uint4*)(pp + koff + 16));
    for (int vt = 0; vt < 8; ++vt) {
      const unsigned short* vv = vp + (size_t)(vt * 16 + idx) * T_DIM + kc + koff;
      v16bf vb = frag_from(*(const uint4*)vv, *(const uint4*)(vv + 16));
      acc[vt] = __builtin_amdgcn_wmma_f32_16x16x32_bf16(false, pa, false, vb,
                                                        (short)0, acc[vt], false, false);
    }
  }

  for (int j = 0; j < 8; ++j) {
    const int trow = q0 + hi * 8 + j;
    const float invl = 1.0f / lrow[j];
    unsigned short* yrow = Y + (size_t)(b * T_DIM + trow) * (H_NUM * VDIM) + h * VDIM;
    for (int vt = 0; vt < 8; ++vt)
      yrow[vt * 16 + idx] = f32_to_bf16(acc[vt][j] * invl);
  }
}

// ---------------------------------------------------------------------------
// Host orchestration
// ---------------------------------------------------------------------------
extern "C" void kernel_launch(void* const* d_in, const int* in_sizes, int n_in,
                              void* d_out, int out_size, void* d_ws, size_t ws_size,
                              hipStream_t stream) {
  const float* x    = (const float*)d_in[0];
  const float* cosv = (const float*)d_in[1];
  const float* sinv = (const float*)d_in[2];
  const float* wqd  = (const float*)d_in[3];
  const float* wqu  = (const float*)d_in[4];
  const float* wkvd = (const float*)d_in[5];
  const float* wkvu = (const float*)d_in[6];
  const float* wo   = (const float*)d_in[7];
  float* out = (float*)d_out;

  char* ws = (char*)d_ws;
  size_t off = 0;
  auto alloc = [&](size_t bytes) -> void* {
    void* p = ws + off;
    off = (off + bytes + 255) & ~(size_t)255;
    return p;
  };

  unsigned short* xb    = (unsigned short*)alloc((size_t)MTOT * C_DIM * 2);
  unsigned short* wqdb  = (unsigned short*)alloc((size_t)QRANK * C_DIM * 2);
  unsigned short* wqub  = (unsigned short*)alloc((size_t)(H_NUM * HEADD) * QRANK * 2);
  unsigned short* wkvdb = (unsigned short*)alloc((size_t)KVRANK * C_DIM * 2);
  unsigned short* wkvub = (unsigned short*)alloc((size_t)(H_NUM * 256) * KVRANK * 2);
  unsigned short* wob   = (unsigned short*)alloc((size_t)C_DIM * (H_NUM * VDIM) * 2);
  unsigned short* qlat  = (unsigned short*)alloc((size_t)MTOT * QRANK * 2);
  unsigned short* kvlat = (unsigned short*)alloc((size_t)MTOT * KVRANK * 2);
  unsigned short* qraw  = (unsigned short*)alloc((size_t)MTOT * (H_NUM * HEADD) * 2);
  unsigned short* kvraw = (unsigned short*)alloc((size_t)MTOT * (H_NUM * 256) * 2);
  unsigned short* Qa    = (unsigned short*)alloc((size_t)B_DIM * H_NUM * T_DIM * HEADD * 2);
  unsigned short* Ka    = (unsigned short*)alloc((size_t)B_DIM * H_NUM * T_DIM * HEADD * 2);
  unsigned short* Vt    = (unsigned short*)alloc((size_t)B_DIM * H_NUM * 128 * T_DIM * 2);
  unsigned short* Yb    = (unsigned short*)alloc((size_t)MTOT * (H_NUM * VDIM) * 2);

  auto cvt = [&](const float* src, unsigned short* dst, size_t n) {
    cvt_f32_bf16<<<(int)((n + 255) / 256), 256, 0, stream>>>(src, dst, (int)n);
  };
  cvt(x,    xb,    (size_t)MTOT * C_DIM);
  cvt(wqd,  wqdb,  (size_t)QRANK * C_DIM);
  cvt(wqu,  wqub,  (size_t)(H_NUM * HEADD) * QRANK);
  cvt(wkvd, wkvdb, (size_t)KVRANK * C_DIM);
  cvt(wkvu, wkvub, (size_t)(H_NUM * 256) * KVRANK);
  cvt(wo,   wob,   (size_t)C_DIM * (H_NUM * VDIM));

  auto gemm = [&](const unsigned short* A, const unsigned short* W,
                  float* Cf, unsigned short* Cb, int M, int N, int K) {
    dim3 grid(N / 64, M / 128);
    gemm_bf16_wmma<<<grid, 256, 0, stream>>>(A, W, Cf, Cb, M, N, K);
  };

  // q = x @ wq_down^T @ wq_up^T
  gemm(xb,   wqdb, nullptr, qlat, MTOT, QRANK, C_DIM);
  gemm(qlat, wqub, nullptr, qraw, MTOT, H_NUM * HEADD, QRANK);
  // kv = x @ wkv_down^T @ wkv_up^T
  gemm(xb,    wkvdb, nullptr, kvlat, MTOT, KVRANK, C_DIM);
  gemm(kvlat, wkvub, nullptr, kvraw, MTOT, H_NUM * 256, KVRANK);

  // RoPE + RMS-norm + K/V split  (one wave per (b,t,h))
  rope_rms_split<<<(B_DIM * T_DIM * H_NUM) / 8, 256, 0, stream>>>(
      qraw, kvraw, cosv, sinv, Qa, Ka, Vt);

  // causal attention
  dim3 agrid(T_DIM / 64, B_DIM * H_NUM);
  mla_flash_attn<<<agrid, 128, 0, stream>>>(Qa, Ka, Vt, Yb);

  // out = y @ wo^T  (f32 straight into d_out)
  gemm(Yb, wob, out, nullptr, MTOT, C_DIM, H_NUM * VDIM);
}
